// _PositionAttentionModule_18691697672452
// MI455X (gfx1250) — matrix operand order repaired and checked by reading, not compiled, not run on hardware
//
#include <hip/hip_runtime.h>
#include <hip/hip_bf16.h>

// ---------------------------------------------------------------------------
// PositionAttentionModule (DANet PAM), B=8, C=256, CR=32, H=W=64, N=4096.
//   q = Wb x + bb ; k = Wc x + bc ; v = Wd x + bd        (1x1 convs)
//   energy = q^T k ; attn = softmax_j(energy)
//   out = alpha * (v attn^T) + x
//
// MI455X (gfx1250) strategy:
//  - Fused attention per 16-query tile; the 16x4096 score row never leaves
//    the WGP: fp32 scores (256KB) + staged bf16 P (128KB, overlapped) use
//    exactly the 320KB LDS.
//  - Both big matmuls are v_wmma_f32_16x16x32_bf16 (K=32 == CR exactly).
//  - Softmax->bf16 conversion happens ONCE per element in phase 2; phase 3's
//    inner loop is pure ds_load_b128 / global_load_b128 / v_wmma.
// ---------------------------------------------------------------------------

typedef __attribute__((ext_vector_type(16))) __bf16 v16bf;
typedef __attribute__((ext_vector_type(8)))  __bf16 bf16x8;
typedef __attribute__((ext_vector_type(8)))  float  v8f;

#define BATCH 8
#define CH    256
#define CRED  32
#define NPIX  4096

static __device__ inline __bf16 f2bf(float f) {
    unsigned u = __builtin_bit_cast(unsigned, f);
    u += 0x7FFFu + ((u >> 16) & 1u);           // round-to-nearest-even
    unsigned short h = (unsigned short)(u >> 16);
    return __builtin_bit_cast(__bf16, h);
}

// ---------------------------------------------------------------------------
// Kernel 1: projections.  Grid (N/64, B), 256 threads, 64KB dynamic LDS.
// Emits:
//   qT [B][N][32]  bf16  (row i contiguous over reduced channels)
//   kT [B][N][32]  bf16
//   vW [B][256][N] bf16  (row c contiguous over pixels)
// ---------------------------------------------------------------------------
__global__ void pam_proj_kernel(const float* __restrict__ x,
                                const float* __restrict__ Wb, const float* __restrict__ bb,
                                const float* __restrict__ Wc, const float* __restrict__ bc,
                                const float* __restrict__ Wd, const float* __restrict__ bd,
                                __bf16* __restrict__ qT, __bf16* __restrict__ kT,
                                __bf16* __restrict__ vW) {
    extern __shared__ float xs[];             // [256 channels][64 pixels]
    const int b   = blockIdx.y;
    const int p0  = blockIdx.x * 64;
    const int tid = threadIdx.x;

    const float* xb = x + (size_t)b * CH * NPIX;
    for (int idx = tid; idx < CH * 64; idx += 256) {
        const int c = idx >> 6, p = idx & 63;
        xs[idx] = xb[(size_t)c * NPIX + p0 + p];
    }
    __syncthreads();

    const int p  = tid & 63;
    const int og = tid >> 6;                  // 4 output groups
    for (int o = og; o < CRED + CRED + CH; o += 4) {
        const float* wrow;
        float bias;
        if (o < CRED)            { wrow = Wb + (size_t)o * CH;          bias = bb[o]; }
        else if (o < 2 * CRED)   { wrow = Wc + (size_t)(o - CRED) * CH; bias = bc[o - CRED]; }
        else                     { wrow = Wd + (size_t)(o - 64) * CH;   bias = bd[o - 64]; }
        float s = bias;
#pragma unroll 8
        for (int c = 0; c < CH; ++c) s += wrow[c] * xs[c * 64 + p];
        const __bf16 h = f2bf(s);
        if (o < CRED)            qT[((size_t)b * NPIX + p0 + p) * CRED + o] = h;
        else if (o < 2 * CRED)   kT[((size_t)b * NPIX + p0 + p) * CRED + (o - CRED)] = h;
        else                     vW[((size_t)b * CH + (o - 64)) * NPIX + p0 + p] = h;
    }
}

// ---------------------------------------------------------------------------
// Kernel 2: fused attention.  Grid (N/16, B), 256 threads (8 waves),
// dynamic LDS = 320KB exactly:
//   region A: bytes [0, 256K)     = S fp32 [16][4096]
//   region B: bytes [256K, 320K)  = reduction scratch, then P bf16 rows 0..7
//   After conversion: P bf16 rows 8..15 live at bytes [0, 64K).
// ---------------------------------------------------------------------------
#define ATTN_SMEM_BYTES (16 * NPIX * 4 + 16 * NPIX * 2 / 2)   // 262144 + 65536

__global__ void pam_attn_kernel(const __bf16* __restrict__ qT,
                                const __bf16* __restrict__ kT,
                                const __bf16* __restrict__ vW,
                                const float* __restrict__ x,
                                const float* __restrict__ alphaPtr,
                                float* __restrict__ out) {
    extern __shared__ float smem[];
    float*  S      = smem;                     // [16][4096] fp32 scores
    float*  red    = smem + 16 * NPIX;         // [16][16] scratch (region B)
    float*  rowmax = red + 256;                // [16]
    float*  rowinv = rowmax + 16;              // [16]
    __bf16* Pb     = (__bf16*)smem;            // bf16 view of all of LDS

    const int b    = blockIdx.y;
    const int i0   = blockIdx.x * 16;          // query tile
    const int tid  = threadIdx.x;
    const int wave = tid >> 5;
    const int lane = tid & 31;
    const int r    = lane & 15;                // A-row / B-col / C-col index
    const int hi   = lane >> 4;                // lane half

    // ---- Load Q tile as WMMA A operand (16x32 bf16, contiguous k = 16*hi + e) -------
    const __bf16* qrow = qT + ((size_t)b * NPIX + i0 + r) * CRED;
    bf16x8 qlo = *(const bf16x8*)(qrow + hi * 16);       // K = hi*16 .. +7
    bf16x8 qhi = *(const bf16x8*)(qrow + hi * 16 + 8);   // K = hi*16+8 .. +7
    v16bf aq;
#pragma unroll
    for (int e = 0; e < 8; ++e) { aq[e] = qlo[e]; aq[e + 8] = qhi[e]; }

    // ---- Phase 1: S = Q^T K, one WMMA per 16x16 key tile (K dim = 32) ------
    const __bf16* kbase = kT + (size_t)b * NPIX * CRED;
    for (int jt = wave; jt < NPIX / 16; jt += 8) {
        const int j0 = jt * 16;
        const __bf16* kp = kbase + (size_t)(j0 + r) * CRED + hi * 16;
        bf16x8 b0 = ((const bf16x8*)kp)[0];
        bf16x8 b1 = ((const bf16x8*)kp)[1];
        v16bf bk;
#pragma unroll
        for (int e = 0; e < 8; ++e) { bk[e] = b0[e]; bk[e + 8] = b1[e]; }
        v8f c = {};
        c = __builtin_amdgcn_wmma_f32_16x16x32_bf16(false, aq, false, bk,
                                                    (short)0, c, false, false);
#pragma unroll
        for (int rr = 0; rr < 8; ++rr)
            S[(rr + 8 * hi) * NPIX + j0 + r] = c[rr];
    }
    __syncthreads();

    // ---- Phase 2a: row max & sum (16 threads per row, no writeback) --------
    {
        const int row = tid >> 4;              // 0..15
        const int t   = tid & 15;              // 0..15
        const float* srow = S + row * NPIX;
        float lmax = -3.402823466e+38f;
        for (int c2 = t; c2 < NPIX; c2 += 16) lmax = fmaxf(lmax, srow[c2]);
        red[row * 16 + t] = lmax;
        __syncthreads();
        if (t == 0) {
            float m = red[row * 16];
#pragma unroll
            for (int u = 1; u < 16; ++u) m = fmaxf(m, red[row * 16 + u]);
            rowmax[row] = m;
        }
        __syncthreads();
        const float m = rowmax[row];
        float lsum = 0.f;
        for (int c2 = t; c2 < NPIX; c2 += 16) lsum += __expf(srow[c2] - m);
        red[row * 16 + t] = lsum;
        __syncthreads();
        if (t == 0) {
            float sacc = 0.f;
#pragma unroll
            for (int u = 0; u < 16; ++u) sacc += red[row * 16 + u];
            rowinv[row] = 1.0f / sacc;
        }
        __syncthreads();
    }

    // ---- Phase 2b: staged fp32 -> normalized bf16 conversion ---------------
    // Hoist per-row stats to registers (scratch gets clobbered by stage 1).
    const int   crow = tid >> 5;               // 0..7 (one wave per row)
    const int   tc   = tid & 31;
    const float m0 = rowmax[crow],     n0 = rowinv[crow];
    const float m1 = rowmax[crow + 8], n1 = rowinv[crow + 8];
    __syncthreads();
    // Stage 1: rows 0..7 -> region B (bf16 elements 131072 + row*4096)
    {
        const float* src = S + crow * NPIX;
        __bf16*      dst = Pb + 131072 + crow * NPIX;
        for (int ch = tc; ch < NPIX / 8; ch += 32) {
            const int e0 = ch * 8;
            const float4 fa = *(const float4*)(src + e0);
            const float4 fb = *(const float4*)(src + e0 + 4);
            bf16x8 pk;
            pk[0] = f2bf(__expf(fa.x - m0) * n0);
            pk[1] = f2bf(__expf(fa.y - m0) * n0);
            pk[2] = f2bf(__expf(fa.z - m0) * n0);
            pk[3] = f2bf(__expf(fa.w - m0) * n0);
            pk[4] = f2bf(__expf(fb.x - m0) * n0);
            pk[5] = f2bf(__expf(fb.y - m0) * n0);
            pk[6] = f2bf(__expf(fb.z - m0) * n0);
            pk[7] = f2bf(__expf(fb.w - m0) * n0);
            *(bf16x8*)(dst + e0) = pk;
        }
    }
    __syncthreads();
    // Stage 2: rows 8..15 -> bytes [0,64K) (fp32 rows 0..3 are dead now)
    {
        const float* src = S + (crow + 8) * NPIX;
        __bf16*      dst = Pb + crow * NPIX;
        for (int ch = tc; ch < NPIX / 8; ch += 32) {
            const int e0 = ch * 8;
            const float4 fa = *(const float4*)(src + e0);
            const float4 fb = *(const float4*)(src + e0 + 4);
            bf16x8 pk;
            pk[0] = f2bf(__expf(fa.x - m1) * n1);
            pk[1] = f2bf(__expf(fa.y - m1) * n1);
            pk[2] = f2bf(__expf(fa.z - m1) * n1);
            pk[3] = f2bf(__expf(fa.w - m1) * n1);
            pk[4] = f2bf(__expf(fb.x - m1) * n1);
            pk[5] = f2bf(__expf(fb.y - m1) * n1);
            pk[6] = f2bf(__expf(fb.z - m1) * n1);
            pk[7] = f2bf(__expf(fb.w - m1) * n1);
            *(bf16x8*)(dst + e0) = pk;
        }
    }
    __syncthreads();

    // ---- Phase 3: feat tile = P * V^T  (wave handles 2 channel tiles) ------
    const float alpha = alphaPtr[0];
    // Split row base for P: rows 0..7 in region B, rows 8..15 at offset 0.
    const __bf16* prow = Pb + ((r < 8) ? (131072 + r * NPIX) : ((r - 8) * NPIX));
    const __bf16* vB   = vW + (size_t)b * CH * NPIX;

    for (int ct = 0; ct < 2; ++ct) {
        const int c0 = (wave + ct * 8) * 16;
        const __bf16* vrow = vB + (size_t)(c0 + r) * NPIX + hi * 16;
        v8f acc = {};
#pragma unroll 2
        for (int jt2 = 0; jt2 < NPIX / 32; ++jt2) {
            const int jb = jt2 * 32;
            // A = softmax tile (16 x 32) bf16, straight from LDS
            bf16x8 a0 = *(const bf16x8*)(prow + jb + hi * 16);
            bf16x8 a1 = *(const bf16x8*)(prow + jb + hi * 16 + 8);
            v16bf ap;
#pragma unroll
            for (int e = 0; e < 8; ++e) { ap[e] = a0[e]; ap[e + 8] = a1[e]; }
            // B = V^T tile (32 x 16): lane col = c0+r, contiguous j run
            const bf16x8* vp = (const bf16x8*)(vrow + jb);
            bf16x8 b0 = vp[0], b1 = vp[1];
            v16bf bv;
#pragma unroll
            for (int e = 0; e < 8; ++e) { bv[e] = b0[e]; bv[e + 8] = b1[e]; }
            acc = __builtin_amdgcn_wmma_f32_16x16x32_bf16(false, ap, false, bv,
                                                          (short)0, acc, false, false);
        }
        // out[b, c, i] = alpha * feat + x
#pragma unroll
        for (int rr = 0; rr < 8; ++rr) {
            const int    i   = i0 + rr + 8 * hi;
            const int    c   = c0 + r;
            const size_t idx = ((size_t)b * CH + c) * NPIX + i;
            out[idx] = alpha * acc[rr] + x[idx];
        }
    }
}

// ---------------------------------------------------------------------------
extern "C" void kernel_launch(void* const* d_in, const int* in_sizes, int n_in,
                              void* d_out, int out_size, void* d_ws, size_t ws_size,
                              hipStream_t stream) {
    const float* x     = (const float*)d_in[0];
    const float* Wb    = (const float*)d_in[1];
    const float* bb    = (const float*)d_in[2];
    const float* Wc    = (const float*)d_in[3];
    const float* bc    = (const float*)d_in[4];
    const float* Wd    = (const float*)d_in[5];
    const float* bd    = (const float*)d_in[6];
    const float* alpha = (const float*)d_in[7];
    float*       out   = (float*)d_out;

    // Workspace carve (bf16): qT 2MB | kT 2MB | vW 16MB  = 20MB total
    char* ws = (char*)d_ws;
    __bf16* qT = (__bf16*)(ws);
    __bf16* kT = (__bf16*)(ws + (size_t)2 * 1024 * 1024);
    __bf16* vW = (__bf16*)(ws + (size_t)4 * 1024 * 1024);

    pam_proj_kernel<<<dim3(NPIX / 64, BATCH), 256, 256 * 64 * sizeof(float), stream>>>(
        x, Wb, bb, Wc, bc, Wd, bd, qT, kT, vW);

    pam_attn_kernel<<<dim3(NPIX / 16, BATCH), 256, ATTN_SMEM_BYTES, stream>>>(
        qT, kT, vW, x, alpha, out);
}
